// SlidingWindowEncoder_26027501814331
// MI455X (gfx1250) — matrix operand order repaired and checked by reading, not compiled, not run on hardware
//
#include <hip/hip_runtime.h>
#include <hip/hip_bf16.h>

typedef __attribute__((ext_vector_type(16))) _Float16 v16h;
typedef __attribute__((ext_vector_type(8)))  _Float16 v8h;
typedef __attribute__((ext_vector_type(8)))  float    v8f;

#define B_   8
#define L_   16384
#define D_   256
#define H_   8
#define HD_  32
#define DFF_ 1024
#define WS_  32
#define NW_  (L_ / WS_)

#define TOK_ 64                        // tokens per block = 2 windows
#define MT_  (TOK_ / 16)               // 4 row-tiles

// LDS strides (elements)
#define SX_STR  264   // 64x256 f16 tile, row = 528B = 33*16  (16B aligned rows)
#define SVT_STR 72    // V^T 256x64 f16, row = 144B = 9*16    (16B aligned rows)
#define SS_STR  33    // scores f32 (scalar access, conflict pad)
#define SF_STR  257   // f32 scratch (bank-conflict-free across rows)
#define SH_STR  1040  // 64x1024 f16 hidden, row = 2080B = 130*16

// LDS byte offsets (hand-packed, phase-aliased)
// sX:   [0, 33792)                         input q (f16); later LN1 output x
// union [33792, 222208):
//   sKV @ 33792  (33792)                   kv input; later attention O
//   sQ  @ 67584  (33792)
//   sK  @ 101376 (33792)
//   sVt @ 135168 (36864)                   V^T [feature][token]
//   sS  @ 172032 (8 waves * 32*33*4 = 33792)  per-wave score tile (reused per window)
//   sP  @ 205824 (8 waves * 32*32*2 = 16384)  per-wave probs (reused per window)
//   sH  @ 33792  (64*1040*2 = 133120)      FFN hidden; aliases dead attention bufs
// sF32: [222208, 288000)  (64*257*4 = 65792)  post-GEMM f32 for residual+LN
#define OFF_SX   0
#define OFF_SKV  33792
#define OFF_SQ   67584
#define OFF_SK   101376
#define OFF_SVT  135168
#define OFF_SS   172032
#define OFF_SP   205824
#define OFF_SH   33792
#define OFF_SF32 222208
#define SMEM_BYTES 288000

// ---------------- WMMA fragment helpers (CDNA5 16x16x32 f16) ----------------

// A-matrix 16x32 f16: lane L -> row M = L%16; K chunks [8*(L/16), +8) and [16+8*(L/16), +8)
__device__ __forceinline__ v16h fragA(const _Float16* base, int row0, int stride,
                                      int k0, int lane) {
  const int m  = lane & 15;
  const int hw = lane >> 4;
  const _Float16* p = base + (row0 + m) * stride + k0 + hw * 8;
  v8h lo = *(const v8h*)p;
  v8h hi = *(const v8h*)(p + 16);
  v16h r;
#pragma unroll
  for (int i = 0; i < 8; ++i) { r[i] = lo[i]; r[i + 8] = hi[i]; }
  return r;
}

// B-matrix 32x16 f16: lane L -> col N = L%16; K chunks [8*(L/16), +8) and [16+8*(L/16), +8)
// Source is stored as B^T rows (N x K row-major), so each chunk is contiguous.
__device__ __forceinline__ v16h fragB(const _Float16* base, int n0, int stride,
                                      int k0, int lane) {
  const int n  = lane & 15;
  const int hw = lane >> 4;
  const _Float16* p = base + (n0 + n) * stride + k0 + hw * 8;
  v8h lo = *(const v8h*)p;
  v8h hi = *(const v8h*)(p + 16);
  v16h r;
#pragma unroll
  for (int i = 0; i < 8; ++i) { r[i] = lo[i]; r[i + 8] = hi[i]; }
  return r;
}

__device__ __forceinline__ v8f wmma32(v16h a, v16h b, v8f c) {
  return __builtin_amdgcn_wmma_f32_16x16x32_f16(false, a, false, b, (short)0, c,
                                                false, false);
}

// C/D 16x16 f32 layout: VGPR i -> row M = i + 8*(lane/16); col N = lane%16
__device__ __forceinline__ void storeC_f16(v8f c, _Float16* out, int r0, int c0,
                                           int stride, int lane, bool relu) {
  const int n  = lane & 15;
  const int rb = r0 + ((lane >> 4) << 3);
#pragma unroll
  for (int i = 0; i < 8; ++i) {
    float v = c[i];
    if (relu) v = v > 0.f ? v : 0.f;
    out[(rb + i) * stride + c0 + n] = (_Float16)v;
  }
}

__device__ __forceinline__ void storeC_f16T(v8f c, _Float16* out, int r0, int c0,
                                            int stride, int lane) {
  const int n  = lane & 15;
  const int rb = r0 + ((lane >> 4) << 3);
#pragma unroll
  for (int i = 0; i < 8; ++i)
    out[(c0 + n) * stride + rb + i] = (_Float16)c[i];
}

__device__ __forceinline__ void storeC_f32(v8f c, float* out, int r0, int c0,
                                           int stride, int lane) {
  const int n  = lane & 15;
  const int rb = r0 + ((lane >> 4) << 3);
#pragma unroll
  for (int i = 0; i < 8; ++i)
    out[(rb + i) * stride + c0 + n] = c[i];
}

// MTx32 output tile block: A (16*MT x KDIM f16, LDS) @ W^T (N x KDIM f16, global) + bias.
// Each B-fragment pair is reused across MT row-tiles -> 2*MT WMMAs per k-step.
// k-loop unroll capped at 2: keeps the wave under 256 VGPRs (no s_set_vgpr_msb
// churn, 2 waves/SIMD schedulable) while still double-buffering the load clauses.
template <int KDIM, int MT>
__device__ __forceinline__ void gemmMx32(const _Float16* sA, int strA,
                                         const _Float16* wT, const float* bias,
                                         int n0, int lane, v8f acc[MT][2]) {
#pragma unroll
  for (int m = 0; m < MT; ++m) { acc[m][0] = (v8f){}; acc[m][1] = (v8f){}; }
#pragma unroll 2
  for (int k0 = 0; k0 < KDIM; k0 += 32) {
    v16h b0 = fragB(wT, n0, KDIM, k0, lane);
    v16h b1 = fragB(wT, n0 + 16, KDIM, k0, lane);
#pragma unroll
    for (int m = 0; m < MT; ++m) {
      v16h a = fragA(sA, m * 16, strA, k0, lane);
      acc[m][0] = wmma32(a, b0, acc[m][0]);
      acc[m][1] = wmma32(a, b1, acc[m][1]);
    }
  }
  const int n = lane & 15;
  const float bb0 = bias[n0 + n];
  const float bb1 = bias[n0 + 16 + n];
#pragma unroll
  for (int m = 0; m < MT; ++m) {
#pragma unroll
    for (int i = 0; i < 8; ++i) { acc[m][0][i] += bb0; acc[m][1][i] += bb1; }
  }
}

// ---------------- weight f32 -> f16 transpose (W is KxN, out is NxK) ----------------

__global__ void wconv_kernel(const float* __restrict__ in, _Float16* __restrict__ out,
                             int K, int N) {
  int idx = blockIdx.x * blockDim.x + threadIdx.x;
  if (idx >= K * N) return;
  int n = idx / K, k = idx - n * K;
  out[idx] = (_Float16)in[k * N + n];
}

// ---------------- fused encoder layer: one block = 2 windows = 64 tokens ----------------

__global__ __launch_bounds__(256, 1)
void encoder_layer_kernel(const float* qin, const float* kvin, float* out,
                          const _Float16* __restrict__ wqT, const _Float16* __restrict__ wkT,
                          const _Float16* __restrict__ wvT, const _Float16* __restrict__ woT,
                          const _Float16* __restrict__ w1T, const _Float16* __restrict__ w2T,
                          const float* __restrict__ bq,  const float* __restrict__ bk,
                          const float* __restrict__ bv,  const float* __restrict__ bo,
                          const float* __restrict__ bf1, const float* __restrict__ bf2,
                          const float* __restrict__ g1,  const float* __restrict__ be1,
                          const float* __restrict__ g2,  const float* __restrict__ be2) {
  __shared__ __align__(16) char smem[SMEM_BYTES];
  _Float16* sX   = (_Float16*)(smem + OFF_SX);
  _Float16* sKV  = (_Float16*)(smem + OFF_SKV);   // later: attention output O
  _Float16* sQ   = (_Float16*)(smem + OFF_SQ);
  _Float16* sK   = (_Float16*)(smem + OFF_SK);
  _Float16* sVt  = (_Float16*)(smem + OFF_SVT);   // V^T: [feature][token]
  float*    sS   = (float*)   (smem + OFF_SS);
  float*    sF32 = (float*)   (smem + OFF_SF32);
  _Float16* sP   = (_Float16*)(smem + OFF_SP);
  _Float16* sH   = (_Float16*)(smem + OFF_SH);

  const int tid  = threadIdx.x;
  const int lane = tid & 31;
  const int w    = tid >> 5;                      // wave id 0..7 (== head id)
  const size_t base = (size_t)blockIdx.x * (TOK_ * D_);

  // keep the weight streams warm in L2 for this block
  __builtin_prefetch(wqT, 0, 1);
  __builtin_prefetch(w1T, 0, 1);

  // ---- load q / kv token rows, f32 -> f16 LDS ----
  for (int i = tid; i < TOK_ * D_ / 4; i += 256) {
    const int r = (i * 4) / D_, c = (i * 4) % D_;
    float4 vq = ((const float4*)(qin + base))[i];
    _Float16* p = sX + r * SX_STR + c;
    p[0] = (_Float16)vq.x; p[1] = (_Float16)vq.y;
    p[2] = (_Float16)vq.z; p[3] = (_Float16)vq.w;
    float4 vk = ((const float4*)(kvin + base))[i];
    _Float16* p2 = sKV + r * SX_STR + c;
    p2[0] = (_Float16)vk.x; p2[1] = (_Float16)vk.y;
    p2[2] = (_Float16)vk.z; p2[3] = (_Float16)vk.w;
  }
  __syncthreads();

  // ---- Q/K/V projections: wave w owns output cols [32w, 32w+32), all 64 rows ----
  {
    const int n0 = w * 32;
    v8f acc[MT_][2];
    gemmMx32<D_, MT_>(sX, SX_STR, wqT, bq, n0, lane, acc);
#pragma unroll
    for (int m = 0; m < MT_; ++m) {
      storeC_f16(acc[m][0], sQ, m * 16, n0, SX_STR, lane, false);
      storeC_f16(acc[m][1], sQ, m * 16, n0 + 16, SX_STR, lane, false);
    }
    gemmMx32<D_, MT_>(sKV, SX_STR, wkT, bk, n0, lane, acc);
#pragma unroll
    for (int m = 0; m < MT_; ++m) {
      storeC_f16(acc[m][0], sK, m * 16, n0, SX_STR, lane, false);
      storeC_f16(acc[m][1], sK, m * 16, n0 + 16, SX_STR, lane, false);
    }
    gemmMx32<D_, MT_>(sKV, SX_STR, wvT, bv, n0, lane, acc);
#pragma unroll
    for (int m = 0; m < MT_; ++m) {
      storeC_f16T(acc[m][0], sVt, m * 16, n0, SVT_STR, lane);
      storeC_f16T(acc[m][1], sVt, m * 16, n0 + 16, SVT_STR, lane);
    }
  }
  __syncthreads();

  // ---- attention: wave w == head w; loop over the 2 windows (block-diagonal) ----
  // Entirely wave-local (scores/probs buffers are per-wave) -> no block barriers inside.
  for (int win = 0; win < 2; ++win) {
    const int o  = win * WS_;                     // token offset of this window
    const int k0 = w * HD_;                       // head's feature slice

    // S = Q_h (32x32) @ K_h^T (32x32), scaled
    {
      v8f c00 = (v8f){}, c01 = (v8f){}, c10 = (v8f){}, c11 = (v8f){};
      v16h a0 = fragA(sQ, o, SX_STR, k0, lane);
      v16h a1 = fragA(sQ, o + 16, SX_STR, k0, lane);
      v16h b0 = fragB(sK, o, SX_STR, k0, lane);   // rows of K == cols of K^T
      v16h b1 = fragB(sK, o + 16, SX_STR, k0, lane);
      c00 = wmma32(a0, b0, c00);
      c01 = wmma32(a0, b1, c01);
      c10 = wmma32(a1, b0, c10);
      c11 = wmma32(a1, b1, c11);
      float* sSw = sS + w * 32 * SS_STR;
      const float scale = 0.17677669529663687f;   // 1/sqrt(32)
      const int n = lane & 15, rb = (lane >> 4) << 3;
#pragma unroll
      for (int i = 0; i < 8; ++i) {
        sSw[(rb + i) * SS_STR + n]           = c00[i] * scale;
        sSw[(rb + i) * SS_STR + 16 + n]      = c01[i] * scale;
        sSw[(16 + rb + i) * SS_STR + n]      = c10[i] * scale;
        sSw[(16 + rb + i) * SS_STR + 16 + n] = c11[i] * scale;
      }
    }

    // softmax: lane = query row within this window
    {
      float* sSw = sS + w * 32 * SS_STR;
      _Float16* sPw = sP + w * 32 * 32;
      float rv[32];
      float m = -1e30f;
#pragma unroll
      for (int j = 0; j < 32; ++j) { rv[j] = sSw[lane * SS_STR + j]; m = fmaxf(m, rv[j]); }
      float s = 0.f;
#pragma unroll
      for (int j = 0; j < 32; ++j) { rv[j] = __expf(rv[j] - m); s += rv[j]; }
      const float inv = 1.f / s;
#pragma unroll
      for (int j = 0; j < 32; ++j) sPw[lane * 32 + j] = (_Float16)(rv[j] * inv);
    }

    // O_h = P (32x32) @ V_h (32x32); write into sKV (kv tile is dead)
    {
      _Float16* sPw = sP + w * 32 * 32;
      const int n0 = w * 32;
      v8f c00 = (v8f){}, c01 = (v8f){}, c10 = (v8f){}, c11 = (v8f){};
      v16h a0 = fragA(sPw, 0, 32, 0, lane);
      v16h a1 = fragA(sPw, 16, 32, 0, lane);
      v16h b0 = fragB(sVt, n0, SVT_STR, o, lane); // rows of V^T == cols of V
      v16h b1 = fragB(sVt, n0 + 16, SVT_STR, o, lane);
      c00 = wmma32(a0, b0, c00);
      c01 = wmma32(a0, b1, c01);
      c10 = wmma32(a1, b0, c10);
      c11 = wmma32(a1, b1, c11);
      storeC_f16(c00, sKV, o, n0, SX_STR, lane, false);
      storeC_f16(c01, sKV, o, n0 + 16, SX_STR, lane, false);
      storeC_f16(c10, sKV, o + 16, n0, SX_STR, lane, false);
      storeC_f16(c11, sKV, o + 16, n0 + 16, SX_STR, lane, false);
    }
  }
  __syncthreads();

  // ---- output projection: sF32 = O @ Wo^T + bo ----
  {
    const int n0 = w * 32;
    v8f acc[MT_][2];
    gemmMx32<D_, MT_>(sKV, SX_STR, woT, bo, n0, lane, acc);
#pragma unroll
    for (int m = 0; m < MT_; ++m) {
      storeC_f32(acc[m][0], sF32, m * 16, n0, SF_STR, lane);
      storeC_f32(acc[m][1], sF32, m * 16, n0 + 16, SF_STR, lane);
    }
  }
  __syncthreads();

  // ---- residual + LayerNorm 1: x = LN(q + o); write f16 into sX ----
  if (tid < TOK_) {
    float* f = sF32 + tid * SF_STR;
    _Float16* x = sX + tid * SX_STR;
    float sum = 0.f, sq = 0.f;
    for (int j = 0; j < D_; ++j) {
      float v = f[j] + (float)x[j];
      sum += v; sq += v * v;
    }
    const float mean = sum * (1.f / D_);
    const float var  = sq * (1.f / D_) - mean * mean;
    const float inv  = rsqrtf(var + 1e-5f);
    for (int j = 0; j < D_; ++j) {
      float v = f[j] + (float)x[j];
      x[j] = (_Float16)((v - mean) * inv * g1[j] + be1[j]);
    }
  }
  __syncthreads();

  // ---- FFN1: sH = relu(x @ W1^T + bf1); wave w owns cols [128w, 128w+128) ----
  for (int t = 0; t < 4; ++t) {
    const int n0 = w * 128 + t * 32;
    v8f acc[MT_][2];
    gemmMx32<D_, MT_>(sX, SX_STR, w1T, bf1, n0, lane, acc);
#pragma unroll
    for (int m = 0; m < MT_; ++m) {
      storeC_f16(acc[m][0], sH, m * 16, n0, SH_STR, lane, true);
      storeC_f16(acc[m][1], sH, m * 16, n0 + 16, SH_STR, lane, true);
    }
  }
  __syncthreads();

  // ---- FFN2: sF32 = h @ W2^T + bf2 ----
  {
    const int n0 = w * 32;
    v8f acc[MT_][2];
    gemmMx32<DFF_, MT_>(sH, SH_STR, w2T, bf2, n0, lane, acc);
#pragma unroll
    for (int m = 0; m < MT_; ++m) {
      storeC_f32(acc[m][0], sF32, m * 16, n0, SF_STR, lane);
      storeC_f32(acc[m][1], sF32, m * 16, n0 + 16, SF_STR, lane);
    }
  }
  __syncthreads();

  // ---- residual + LayerNorm 2, f32 write to global ----
  if (tid < TOK_) {
    float* f = sF32 + tid * SF_STR;
    _Float16* x = sX + tid * SX_STR;
    float sum = 0.f, sq = 0.f;
    for (int j = 0; j < D_; ++j) {
      float v = f[j] + (float)x[j];
      sum += v; sq += v * v;
    }
    const float mean = sum * (1.f / D_);
    const float var  = sq * (1.f / D_) - mean * mean;
    const float inv  = rsqrtf(var + 1e-5f);
    float* o = out + base + (size_t)tid * D_;
    for (int j = 0; j < D_; ++j) {
      float v = f[j] + (float)x[j];
      o[j] = (v - mean) * inv * g2[j] + be2[j];
    }
  }
}

// ---------------- launch ----------------
// roll(±WS/2) along the *window* axis commutes with a purely window-local layer,
// so the reference == layer(layer(q, kv), self) applied independently per window.

extern "C" void kernel_launch(void* const* d_in, const int* in_sizes, int n_in,
                              void* d_out, int out_size, void* d_ws, size_t ws_size,
                              hipStream_t stream) {
  (void)in_sizes; (void)n_in; (void)out_size; (void)ws_size;
  const float* qem  = (const float*)d_in[0];
  const float* kvem = (const float*)d_in[1];
  const float* Wq = (const float*)d_in[2];  const float* bq  = (const float*)d_in[3];
  const float* Wk = (const float*)d_in[4];  const float* bk  = (const float*)d_in[5];
  const float* Wv = (const float*)d_in[6];  const float* bv  = (const float*)d_in[7];
  const float* Wo = (const float*)d_in[8];  const float* bo  = (const float*)d_in[9];
  const float* g1 = (const float*)d_in[10]; const float* be1 = (const float*)d_in[11];
  const float* W1 = (const float*)d_in[12]; const float* bf1 = (const float*)d_in[13];
  const float* W2 = (const float*)d_in[14]; const float* bf2 = (const float*)d_in[15];
  const float* g2 = (const float*)d_in[16]; const float* be2 = (const float*)d_in[17];
  float* out = (float*)d_out;

  _Float16* ws  = (_Float16*)d_ws;
  _Float16* wqT = ws;                 // 256*256
  _Float16* wkT = wqT + D_ * D_;
  _Float16* wvT = wkT + D_ * D_;
  _Float16* woT = wvT + D_ * D_;
  _Float16* w1T = woT + D_ * D_;      // 1024*256 (W1^T)
  _Float16* w2T = w1T + D_ * DFF_;    // 256*1024 (W2^T)

  // weights -> f16 transposed (L2-resident, reused by all blocks x 2 layers)
  const int ct = 256;
  wconv_kernel<<<(D_ * D_ + ct - 1) / ct, ct, 0, stream>>>(Wq, wqT, D_, D_);
  wconv_kernel<<<(D_ * D_ + ct - 1) / ct, ct, 0, stream>>>(Wk, wkT, D_, D_);
  wconv_kernel<<<(D_ * D_ + ct - 1) / ct, ct, 0, stream>>>(Wv, wvT, D_, D_);
  wconv_kernel<<<(D_ * D_ + ct - 1) / ct, ct, 0, stream>>>(Wo, woT, D_, D_);
  wconv_kernel<<<(D_ * DFF_ + ct - 1) / ct, ct, 0, stream>>>(W1, w1T, D_, DFF_);
  wconv_kernel<<<(DFF_ * D_ + ct - 1) / ct, ct, 0, stream>>>(W2, w2T, DFF_, D_);

  dim3 grid(B_ * NW_ / 2);   // 2048 blocks, 2 windows each (halves L2 weight traffic)
  dim3 block(256);           // 8 wave32 waves
  // layer 1: q = query_em, kv = key_value_em -> d_out
  encoder_layer_kernel<<<grid, block, 0, stream>>>(qem, kvem, out,
      wqT, wkT, wvT, woT, w1T, w2T, bq, bk, bv, bo, bf1, bf2, g1, be1, g2, be2);
  // layer 2: self-attention (roll is a no-op for a window-local layer); in-place per block
  encoder_layer_kernel<<<grid, block, 0, stream>>>(out, out, out,
      wqT, wkT, wvT, woT, w1T, w2T, bq, bk, bv, bo, bf1, bf2, g1, be1, g2, be2);
}
